// MultiHeadAttention_5282809774859
// MI455X (gfx1250) — matrix operand order repaired and checked by reading, not compiled, not run on hardware
//
#include <hip/hip_runtime.h>

#define B_ 2
#define T_ 2048
#define E_ 768
#define H_ 12
#define DK_ 64
#define LN_EPS 1e-5f

typedef _Float16 v16h __attribute__((ext_vector_type(16)));
typedef float    v8f  __attribute__((ext_vector_type(8)));

union FragU { unsigned int u[8]; v16h v; };

// A-operand fragment (16x32 f16). ISA layout: lane m(0..15)/m+16 holds row m;
// lane-half selects K group {0-7,16-23} vs {8-15,24-31}; each VGPR = 2 halves.
__device__ __forceinline__ v16h load_a_frag(const _Float16* base, int strideHalves, int lane) {
  const unsigned int* p = (const unsigned int*)(base + (lane & 15) * strideHalves);
  const int kh = (lane >> 4) & 1;
  FragU f;
#pragma unroll
  for (int v = 0; v < 4; ++v) {
    f.u[v]     = p[kh * 4 + v];       // K = kh*8 + 2v
    f.u[v + 4] = p[8 + kh * 4 + v];   // K = 16 + kh*8 + 2v
  }
  return f.v;
}

// B-operand fragment (32x16 f16) read from a B-transposed LDS tile [N][K].
// lane-half selects K group {0-7,16-23} vs {8-15,24-31} of column n (same as A).
__device__ __forceinline__ v16h load_b_frag(const _Float16* base, int strideHalves, int lane) {
  const unsigned int* p = (const unsigned int*)(base + (lane & 15) * strideHalves);
  const int kh = (lane >> 4) & 1;
  FragU f;
#pragma unroll
  for (int v = 0; v < 4; ++v) { f.u[v] = p[kh * 4 + v]; f.u[v + 4] = p[8 + kh * 4 + v]; }
  return f.v;
}

__device__ __forceinline__ v8f wmma_f16(v16h a, v16h b, v8f c) {
  return __builtin_amdgcn_wmma_f32_16x16x32_f16(false, a, false, b, (short)0, c, false, false);
}

// CDNA5 async global->LDS copy (ASYNCcnt-tracked), 16 bytes per lane.
// ldsOff = group-segment byte offset (low 32 bits of generic LDS pointer).
__device__ __forceinline__ void async_copy16(unsigned ldsOff, const void* gaddr) {
  asm volatile("global_load_async_to_lds_b128 %0, %1, off"
               :: "v"(ldsOff), "v"(gaddr) : "memory");
}
__device__ __forceinline__ void wait_async() {
  asm volatile("s_wait_asynccnt 0x0" ::: "memory");
}

// ---------------- Kernel 1: fused QKV projection ----------------
// grid (64 mtiles, H, 3 proj), block 128 (4 waves). C tile = 64x64.
// q,k stored [B,H,T,DK]; v stored TRANSPOSED [B,H,DK,T] (async-copy friendly).
__global__ __launch_bounds__(128) void qkv_proj_kernel(
    const float* __restrict__ x,
    const float* __restrict__ Wq, const float* __restrict__ bq,
    const float* __restrict__ Wk, const float* __restrict__ bk,
    const float* __restrict__ Wv, const float* __restrict__ bv,
    _Float16* __restrict__ qh, _Float16* __restrict__ kh, _Float16* __restrict__ vh) {
  const int tid = threadIdx.x, lane = tid & 31, w = tid >> 5;
  const int mtile = blockIdx.x, h = blockIdx.y, p = blockIdx.z;
  const float* W    = (p == 0) ? Wq : (p == 1) ? Wk : Wv;
  const float* bias = ((p == 0) ? bq : (p == 1) ? bk : bv) + h * DK_;
  _Float16*   out   = (p == 0) ? qh : (p == 1) ? kh : vh;
  const float* Wh = W + (size_t)h * E_ * DK_;
  const int m0 = mtile * 64;

  __shared__ __attribute__((aligned(16))) _Float16 sA[64 * 34];
  __shared__ __attribute__((aligned(16))) _Float16 sBT[64 * 34];

  v8f zero = {};
  v8f acc[4] = {zero, zero, zero, zero};

  for (int kk = 0; kk < E_ / 32; ++kk) {
    for (int idx = tid; idx < 64 * 32; idx += 128) {      // A: x[m0..m0+63][kk*32..+31]
      int r = idx >> 5, c = idx & 31;
      sA[r * 34 + c] = (_Float16)x[(size_t)(m0 + r) * E_ + kk * 32 + c];
    }
    for (int idx = tid; idx < 64 * 32; idx += 128) {      // B^T: [d][j] = W[(kk*32+j)*64 + d]
      int j = idx >> 6, d = idx & 63;
      sBT[d * 34 + j] = (_Float16)Wh[(size_t)(kk * 32 + j) * DK_ + d];
    }
    __syncthreads();
    v16h a = load_a_frag(sA + w * 16 * 34, 34, lane);
#pragma unroll
    for (int ns = 0; ns < 4; ++ns) {
      v16h bf = load_b_frag(sBT + ns * 16 * 34, 34, lane);
      acc[ns] = wmma_f16(a, bf, acc[ns]);
    }
    __syncthreads();
  }

  const int hi = (lane >> 4) & 1, nl = lane & 15;
  if (p == 2) {
    // V transposed: vt[((b*H+h)*DK + d)*T + t]; 8 consecutive t per lane -> one 16B store
    const int bb = m0 >> 11;
    const int tbase = (m0 & (T_ - 1)) + w * 16 + 8 * hi;
#pragma unroll
    for (int ns = 0; ns < 4; ++ns) {
      int d = ns * 16 + nl;
      float bval = bias[d];
      union { _Float16 hv[8]; uint4 q; } pk;
#pragma unroll
      for (int r = 0; r < 8; ++r) pk.hv[r] = (_Float16)(acc[ns][r] + bval);
      *(uint4*)(out + ((size_t)(bb * H_ + h) * DK_ + d) * T_ + tbase) = pk.q;
    }
  } else {
#pragma unroll
    for (int ns = 0; ns < 4; ++ns) {
      int d = ns * 16 + nl;
      float bval = bias[d];
#pragma unroll
      for (int r = 0; r < 8; ++r) {
        int m = m0 + w * 16 + r + 8 * hi;
        int bb = m >> 11, t = m & (T_ - 1);
        out[(size_t)((bb * H_ + h) * T_ + t) * DK_ + d] = (_Float16)(acc[ns][r] + bval);
      }
    }
  }
}

// ---------------- Kernel 2: attention (unnormalized flash pass) ----------------
// grid (T/64, H, B), block 128. Double-buffered K / V^T tiles filled by
// global_load_async_to_lds_b128, overlapped with WMMA on the current buffer.
__global__ __launch_bounds__(128) void attn_kernel(
    const _Float16* __restrict__ qh, const _Float16* __restrict__ kh,
    const _Float16* __restrict__ vh, float* __restrict__ zun,
    float* __restrict__ denp) {
  const int tid = threadIdx.x, lane = tid & 31, w = tid >> 5;
  const int tt = blockIdx.x, h = blockIdx.y, b = blockIdx.z;
  const int t0 = tt * 64;
  const _Float16* qb  = qh + (size_t)((b * H_ + h) * T_) * DK_;
  const _Float16* kb  = kh + (size_t)((b * H_ + h) * T_) * DK_;
  const _Float16* vtb = vh + (size_t)((b * H_ + h) * DK_) * T_;  // transposed layout
  float* zb = zun + (size_t)((b * H_ + h) * T_) * DK_;

  // Strides: K tile 72 halves (144B row: 16B aligned, bank step 36 dwords -> conflict-free)
  //          V^T tile 40 halves (80B row: 16B aligned, bank step 20 dwords -> conflict-free)
  __shared__ __attribute__((aligned(16))) _Float16 sQ[64 * 66];
  __shared__ __attribute__((aligned(16))) _Float16 sK2[2][32 * 72];
  __shared__ __attribute__((aligned(16))) _Float16 sVT2[2][64 * 40];
  __shared__ __attribute__((aligned(16))) _Float16 sP[4][16 * 34];
  __shared__ float red[128];

  const unsigned sKoff = (unsigned)(size_t)(void*)&sK2[0][0];
  const unsigned sVoff = (unsigned)(size_t)(void*)&sVT2[0][0];

  auto issue_tile = [&](int bi, int s0n) {
    int c = tid;
#pragma unroll
    for (int ps = 0; ps < 2; ++ps, c += 128) {       // K tile: 32 rows x 128B
      int r = c >> 3, sub = c & 7;
      async_copy16(sKoff + bi * (32 * 144) + r * 144 + sub * 16,
                   (const void*)(kb + (size_t)(s0n + r) * DK_ + sub * 8));
    }
    c = tid;
#pragma unroll
    for (int ps = 0; ps < 2; ++ps, c += 128) {       // V^T tile: 64 rows x 64B
      int d = c >> 2, sub = c & 3;
      async_copy16(sVoff + bi * (64 * 80) + d * 80 + sub * 16,
                   (const void*)(vtb + (size_t)d * T_ + s0n + sub * 8));
    }
  };

  issue_tile(0, 0);                                  // prefetch s-tile 0 (overlaps Q staging)
  { // Q tile resident for whole block: [64 t][64 dk], dword copies
    const unsigned int* qu = (const unsigned int*)qb;
    unsigned int* sQu = (unsigned int*)sQ;
    for (int idx = tid; idx < 64 * 32; idx += 128) {
      int r = idx >> 5, c = idx & 31;
      sQu[r * 33 + c] = qu[(size_t)(t0 + r) * 32 + c];
    }
  }
  wait_async();
  __syncthreads();
  v16h aq0 = load_a_frag(sQ + w * 16 * 66, 66, lane);       // dk 0..31
  v16h aq1 = load_a_frag(sQ + w * 16 * 66 + 32, 66, lane);  // dk 32..63

  v8f zero = {};
  v8f zacc[4] = {zero, zero, zero, zero};
  float dsum = 0.0f;
  const float scale = 0.125f;  // 1/sqrt(DK)
  const int hi = (lane >> 4) & 1, nl = lane & 15;

  for (int si = 0; si < T_ / 32; ++si) {
    const int cur = si & 1;
    if (si + 1 < T_ / 32) issue_tile(cur ^ 1, (si + 1) * 32);  // async prefetch next tile

    const _Float16* sKb = &sK2[cur][0];
    const _Float16* sVb = &sVT2[cur][0];
    _Float16* sPw = sP[w];
#pragma unroll
    for (int sub = 0; sub < 2; ++sub) {                    // two 16-col s subtiles
      v8f sacc = zero;
      v16h bk0 = load_b_frag(sKb + sub * 16 * 72, 72, lane);
      v16h bk1 = load_b_frag(sKb + sub * 16 * 72 + 32, 72, lane);
      sacc = wmma_f16(aq0, bk0, sacc);
      sacc = wmma_f16(aq1, bk1, sacc);
#pragma unroll
      for (int r = 0; r < 8; ++r) {                        // exp + pack P (C->A relayout via LDS)
        float e = __expf(sacc[r] * scale);
        dsum += e;
        sPw[(r + 8 * hi) * 34 + sub * 16 + nl] = (_Float16)e;
      }
    }
    v16h ap = load_a_frag(sPw, 34, lane);                  // P [16 t][32 s]
#pragma unroll
    for (int ns = 0; ns < 4; ++ns) {                       // z += P · V
      v16h bvf = load_b_frag(sVb + ns * 16 * 40, 40, lane);
      zacc[ns] = wmma_f16(ap, bvf, zacc[ns]);
    }
    wait_async();                                          // next tile landed
    __syncthreads();
  }

#pragma unroll
  for (int ns = 0; ns < 4; ++ns)
#pragma unroll
    for (int r = 0; r < 8; ++r)
      zb[(size_t)(t0 + w * 16 + r + 8 * hi) * DK_ + ns * 16 + nl] = zacc[ns][r];

  red[tid] = dsum;
  __syncthreads();
  if (tid == 0) {
    float s = 0.0f;
    for (int i = 0; i < 128; ++i) s += red[i];
    denp[(b * H_ + h) * 32 + tt] = s;
  }
}

// ---------------- Kernel 3: per-head denom -> reciprocal ----------------
__global__ void denom_reduce_kernel(const float* __restrict__ denp, float* __restrict__ invd) {
  int h = threadIdx.x;
  if (h < H_) {
    float s = 0.0f;
    for (int b = 0; b < B_; ++b)
      for (int tt = 0; tt < 32; ++tt)
        s += denp[(b * H_ + h) * 32 + tt];
    invd[h] = 1.0f / s;
  }
}

// ---------------- Kernel 4: output projection + residual + LN partials ----------------
// grid (64 mtiles, 12 ntiles), block 128. y written to d_out.
__global__ __launch_bounds__(128) void oproj_kernel(
    const float* __restrict__ zun, const float* __restrict__ invd,
    const float* __restrict__ Wo, const float* __restrict__ bo,
    const float* __restrict__ x, float* __restrict__ y,
    float2* __restrict__ ypart) {
  const int tid = threadIdx.x, lane = tid & 31, w = tid >> 5;
  const int mtile = blockIdx.x, ntile = blockIdx.y;
  const int m0 = mtile * 64, n0 = ntile * 64;

  __shared__ __attribute__((aligned(16))) _Float16 sA[64 * 34];
  __shared__ __attribute__((aligned(16))) _Float16 sBT[64 * 34];
  __shared__ float sInv[H_];
  __shared__ float reds[128], redq[128];

  if (tid < H_) sInv[tid] = invd[tid];
  __syncthreads();

  v8f zero = {};
  v8f acc[4] = {zero, zero, zero, zero};

  for (int kk = 0; kk < E_ / 32; ++kk) {
    const int he = kk >> 1;                 // head of this 32-wide K chunk
    const int dbase = (kk & 1) * 32;
    const float sc = sInv[he];
    for (int idx = tid; idx < 64 * 32; idx += 128) {  // A = normalized z, cast f16
      int r = idx >> 5, j = idx & 31;
      int m = m0 + r, bb = m >> 11, t = m & (T_ - 1);
      sA[r * 34 + j] =
          (_Float16)(zun[(size_t)((bb * H_ + he) * T_ + t) * DK_ + dbase + j] * sc);
    }
    for (int idx = tid; idx < 64 * 32; idx += 128) {  // B^T: [n][j] = Wo[(kk*32+j)*768 + n0+n]
      int j = idx >> 6, n = idx & 63;
      sBT[n * 34 + j] = (_Float16)Wo[(size_t)(kk * 32 + j) * E_ + n0 + n];
    }
    __syncthreads();
    v16h a = load_a_frag(sA + w * 16 * 34, 34, lane);
#pragma unroll
    for (int ns = 0; ns < 4; ++ns) {
      v16h bf = load_b_frag(sBT + ns * 16 * 34, 34, lane);
      acc[ns] = wmma_f16(a, bf, acc[ns]);
    }
    __syncthreads();
  }

  float lsum = 0.0f, lsq = 0.0f;
  const int hi = (lane >> 4) & 1, nl = lane & 15;
#pragma unroll
  for (int ns = 0; ns < 4; ++ns) {
    int n = n0 + ns * 16 + nl;
    float bv = bo[n];
#pragma unroll
    for (int r = 0; r < 8; ++r) {
      int m = m0 + w * 16 + r + 8 * hi;
      float val = acc[ns][r] + bv + x[(size_t)m * E_ + n];
      y[(size_t)m * E_ + n] = val;
      lsum += val;
      lsq += val * val;
    }
  }
  reds[tid] = lsum; redq[tid] = lsq;
  __syncthreads();
  if (tid == 0) {
    float s1 = 0.0f, s2 = 0.0f;
    for (int i = 0; i < 128; ++i) { s1 += reds[i]; s2 += redq[i]; }
    ypart[mtile * 12 + ntile] = make_float2(s1, s2);
  }
}

// ---------------- Kernel 5: per-batch mean / inv-std ----------------
__global__ void stats_kernel(const float2* __restrict__ ypart, float* __restrict__ mus) {
  int bb = threadIdx.x;
  if (bb < B_) {
    double s1 = 0.0, s2 = 0.0;
    for (int mt = 0; mt < 32; ++mt)
      for (int nt = 0; nt < 12; ++nt) {
        float2 p = ypart[(bb * 32 + mt) * 12 + nt];
        s1 += (double)p.x; s2 += (double)p.y;
      }
    const double n = (double)T_ * (double)E_;
    double mu = s1 / n;
    double var = s2 / n - mu * mu;
    mus[bb] = (float)mu;
    mus[B_ + bb] = (float)(1.0 / sqrt(var + (double)LN_EPS));
  }
}

// ---------------- Kernel 6: in-place LayerNorm over [T,E] per batch ----------------
__global__ void ln_kernel(float* __restrict__ y, const float* __restrict__ mus,
                          const float* __restrict__ lnw, const float* __restrict__ lnb) {
  int i0 = (blockIdx.x * blockDim.x + threadIdx.x) * 4;
#pragma unroll
  for (int j = 0; j < 4; ++j) {
    int i = i0 + j;  // grid sized exactly: B*T*E divisible by 1024
    int bb = i / (T_ * E_);
    int rem = i - bb * (T_ * E_);
    y[i] = (y[i] - mus[bb]) * mus[B_ + bb] * lnw[rem] + lnb[rem];
  }
}

extern "C" void kernel_launch(void* const* d_in, const int* in_sizes, int n_in,
                              void* d_out, int out_size, void* d_ws, size_t ws_size,
                              hipStream_t stream) {
  (void)in_sizes; (void)n_in; (void)out_size; (void)ws_size;
  const float* x   = (const float*)d_in[0];
  const float* Wq  = (const float*)d_in[1];
  const float* bq  = (const float*)d_in[2];
  const float* Wk  = (const float*)d_in[3];
  const float* bk  = (const float*)d_in[4];
  const float* Wv  = (const float*)d_in[5];
  const float* bv  = (const float*)d_in[6];
  const float* Wo  = (const float*)d_in[7];
  const float* bo  = (const float*)d_in[8];
  const float* lnw = (const float*)d_in[9];
  const float* lnb = (const float*)d_in[10];
  float* out = (float*)d_out;

  char* ws = (char*)d_ws;
  const size_t NQ = (size_t)B_ * H_ * T_ * DK_;  // 3,145,728
  _Float16* qh  = (_Float16*)(ws);
  _Float16* kh  = (_Float16*)(ws + NQ * 2);
  _Float16* vh  = (_Float16*)(ws + NQ * 4);      // transposed [B,H,DK,T]
  float*    zun = (float*)(ws + NQ * 6);
  float*    denp = (float*)(ws + NQ * 6 + NQ * 4);   // 768 partials
  float*    invd = denp + 1024;                      // 12 reciprocals
  float2*   ypart = (float2*)(invd + 64);            // 768 (sum,sumsq)
  float*    mus = (float*)(ypart + 1024);            // mu[2], rstd[2]

  qkv_proj_kernel<<<dim3(64, H_, 3), 128, 0, stream>>>(x, Wq, bq, Wk, bk, Wv, bv, qh, kh, vh);
  attn_kernel<<<dim3(T_ / 64, H_, B_), 128, 0, stream>>>(qh, kh, vh, zun, denp);
  denom_reduce_kernel<<<1, 32, 0, stream>>>(denp, invd);
  oproj_kernel<<<dim3(64, 12), 128, 0, stream>>>(zun, invd, Wo, bo, x, out, ypart);
  stats_kernel<<<1, 32, 0, stream>>>(ypart, mus);
  ln_kernel<<<3072, 256, 0, stream>>>(out, mus, lnw, lnb);
}
